// SimpleRNN_80238579024581
// MI455X (gfx1250) — compile-verified, matrix operand order checked by reading, not run on hardware
//
#include <hip/hip_runtime.h>
#include <hip/hip_bf16.h>
#include <cstdint>

// ---------------------------------------------------------------------------
// Stacked tanh RNN on MI455X (gfx1250).  B=64, T=256, H=1024, L=4, OUT=1
//   - bf16 operands, f32 accum via v_wmma_f32_16x16x32_bf16
//   - wavefront layer pipeline: step s -> layer l computes t = s - l
//     (serial chain T+L-1 = 259 steps instead of T*L = 1024)
//   - persistent kernel, 128 WGs x 256 threads, device-wide barrier per step
//   - A operand (activations, shared by all waves of a WG) staged into LDS
//     with CDNA5 async loads (ASYNCcnt) when available; B operand (weights,
//     unique per column block) streams on the VMEM path -> both memory paths
//     feed the WMMA pipe in parallel.
// ---------------------------------------------------------------------------

constexpr int RB = 64;    // batch
constexpr int RT = 256;   // timesteps
constexpr int RH = 1024;  // hidden
constexpr int RL = 4;     // layers
constexpr int NWG = 128;  // persistent grid (4 layers x 32 col-blocks)
constexpr int LBH = RL * RB * RH;
constexpr int BH  = RB * RH;

typedef __bf16 bf16_t;
typedef __attribute__((ext_vector_type(16))) __bf16 v16bf;
typedef __attribute__((ext_vector_type(8)))  __bf16 v8bf;
typedef __attribute__((ext_vector_type(8)))  float  v8f;
typedef __attribute__((ext_vector_type(4)))  int    v4i;

// Builtin's parameter types (revealed by round-2 diagnostic): pointers to
// <4 x i32> in addrspace(1) (global) and addrspace(3) (LDS).
typedef __attribute__((address_space(1))) v4i* gv4i_p;
typedef __attribute__((address_space(3))) v4i* lv4i_p;

// LDS staging: 64 rows x 64 halfs per K-block; pitch 72 halfs = 144 B
// (16B-aligned rows; 36 banks/row stride -> conflict-free 16-row frag reads).
constexpr int LDS_PITCH = 72;
constexpr int LDS_BUF   = 64 * LDS_PITCH;  // halfs per buffer

// --------------------- CDNA5 feature detection -----------------------------
#if defined(__has_builtin)
#  if __has_builtin(__builtin_amdgcn_global_load_async_to_lds_b128)
#    define RNN_ASYNC_LDS 1
#  endif
#endif
#ifndef RNN_ASYNC_LDS
#  define RNN_ASYNC_LDS 0
#endif

__device__ __forceinline__ void wait_async0() {
#if RNN_ASYNC_LDS
#  if __has_builtin(__builtin_amdgcn_s_wait_asynccnt)
  __builtin_amdgcn_s_wait_asynccnt(0);
#  else
  asm volatile("s_wait_asynccnt 0x0" ::: "memory");
#  endif
#endif
}

__device__ __forceinline__ float fast_tanh(float x) {
#if __has_builtin(__builtin_amdgcn_tanhf)
  return __builtin_amdgcn_tanhf(x);   // v_tanh_f32
#else
  return tanhf(x);
#endif
}

// ---------------------------------------------------------------------------
// Stage a 64-row x 64-half A block into LDS. 256 threads x 2 x 16B.
// ---------------------------------------------------------------------------
__device__ __forceinline__ void stage_a(const bf16_t* __restrict__ ga,
                                        size_t stride, bf16_t* lbase, int tid) {
#pragma unroll
  for (int i = 0; i < 2; ++i) {
    const int seg = tid + (i << 8);         // 0..511
    const int row = seg >> 3;               // 64 rows
    const int s   = seg & 7;                // 8 x 16B per row
    const bf16_t* g = ga + (size_t)row * stride + s * 8;
    bf16_t* l = lbase + row * LDS_PITCH + s * 8;
#if RNN_ASYNC_LDS
    __builtin_amdgcn_global_load_async_to_lds_b128((gv4i_p)g, (lv4i_p)l, 0, 0);
#else
    *(v8bf*)l = *(const v8bf*)g;            // global_load_b128 + ds_store_b128
#endif
  }
}

// 16-bit WMMA operand striping (ISA 7.12.2):
//   lane<16  : row=lane,   comps 0..7 = K 0..7,  8..15 = K 16..23
//   lane>=16 : row=lane-16, comps 0..7 = K 8..15, 8..15 = K 24..31
__device__ __forceinline__ v16bf lds_frag(const bf16_t* lbase, int m0, int kc,
                                          int lane) {
  const int r = lane & 15, h = lane >> 4;
  const bf16_t* p = lbase + (m0 + r) * LDS_PITCH + kc + 8 * h;
  v8bf lo = *(const v8bf*)(p);
  v8bf hi = *(const v8bf*)(p + 16);
  return __builtin_shufflevector(lo, hi, 0, 1, 2, 3, 4, 5, 6, 7,
                                         8, 9, 10, 11, 12, 13, 14, 15);
}

__device__ __forceinline__ v16bf glb_frag(const bf16_t* __restrict__ base,
                                          size_t row_stride, int lane) {
  const int r = lane & 15, h = lane >> 4;
  const bf16_t* p = base + (size_t)r * row_stride + 8 * h;
  v8bf lo = *(const v8bf*)(p);
  v8bf hi = *(const v8bf*)(p + 16);
  return __builtin_shufflevector(lo, hi, 0, 1, 2, 3, 4, 5, 6, 7,
                                         8, 9, 10, 11, 12, 13, 14, 15);
}

// Device-wide sense-reversal barrier (NWG resident workgroups).
__device__ __forceinline__ void grid_sync(unsigned* cnt, unsigned* gen) {
  __syncthreads();
  if (threadIdx.x == 0) {
    __threadfence();
    unsigned g = __atomic_load_n(gen, __ATOMIC_RELAXED);
    unsigned prev = __atomic_fetch_add(cnt, 1u, __ATOMIC_ACQ_REL);
    if (prev == NWG - 1) {
      __atomic_store_n(cnt, 0u, __ATOMIC_RELAXED);
      __atomic_fetch_add(gen, 1u, __ATOMIC_RELEASE);
    } else {
      while (__atomic_load_n(gen, __ATOMIC_ACQUIRE) == g)
        __builtin_amdgcn_s_sleep(1);
    }
    __threadfence();
  }
  __syncthreads();
}

// --------------------------- prep kernels ----------------------------------
__global__ void rnn_prep_weights(const float* __restrict__ wih,
                                 const float* __restrict__ whh,
                                 const float* __restrict__ bih,
                                 const float* __restrict__ bhh,
                                 bf16_t* __restrict__ wbih,
                                 bf16_t* __restrict__ wbhh,
                                 float* __restrict__ bias) {
  size_t i = (size_t)blockIdx.x * blockDim.x + threadIdx.x;
  if (i < (size_t)RL * RH * RH) {
    wbih[i] = (bf16_t)wih[i];
    wbhh[i] = (bf16_t)whh[i];
  }
  if (i < (size_t)RL * RH) bias[i] = bih[i] + bhh[i];
}

__global__ void rnn_prep_x(const float* __restrict__ x, bf16_t* __restrict__ xb) {
  size_t i = (size_t)blockIdx.x * blockDim.x + threadIdx.x;
  if (i < (size_t)RB * RT * RH) xb[i] = (bf16_t)x[i];
}

__global__ void rnn_zero_ws(unsigned* __restrict__ hbuf_u,
                            unsigned* __restrict__ sync) {
  size_t i = (size_t)blockIdx.x * blockDim.x + threadIdx.x;
  size_t n = (size_t)2 * LBH * sizeof(bf16_t) / sizeof(unsigned);
  if (i < n) hbuf_u[i] = 0u;
  if (i == 0) { sync[0] = 0u; sync[1] = 0u; }
}

// --------------------------- main persistent kernel ------------------------
__global__ void __launch_bounds__(256, 1)
rnn_persistent(const bf16_t* __restrict__ xb,
               const bf16_t* __restrict__ wih,
               const bf16_t* __restrict__ whh,
               const float*  __restrict__ bias,
               bf16_t* __restrict__ hbuf,
               unsigned* __restrict__ sync) {
  __shared__ bf16_t smem[2 * LDS_BUF];     // 18 KB of 320 KB/WGP

  const int wg    = blockIdx.x;            // 0..127
  const int layer = wg >> 5;               // 4 layers
  const int nblk  = wg & 31;               // 32-column block within layer
  const int tid   = threadIdx.x;
  const int lane  = tid & 31;              // wave32
  const int wave  = tid >> 5;              // 8 waves
  const int m0    = (wave & 3) * 16;       // batch-row tile
  const int col0  = nblk * 32 + (wave >> 2) * 16;  // output-column tile
  const int ncol  = lane & 15;
  const int hhalf = lane >> 4;

  const bf16_t* Wih_l = wih + (size_t)layer * RH * RH;
  const bf16_t* Whh_l = whh + (size_t)layer * RH * RH;
  const float   bcol  = bias[layer * RH + col0 + ncol];

  for (int s = 0; s < RT + RL - 1; ++s) {
    const int t = s - layer;
    if (0 <= t && t < RT) {
      const int pw = s & 1;                // write parity; read parity = 1-pw
      const bf16_t* hprev = hbuf + (size_t)(1 - pw) * LBH + (size_t)layer * BH;
      const bf16_t* inp;
      size_t strideA;
      if (layer == 0) { inp = xb + (size_t)t * RH; strideA = (size_t)RT * RH; }
      else            { inp = hbuf + (size_t)(1 - pw) * LBH +
                              (size_t)(layer - 1) * BH;
                        strideA = RH; }
      bf16_t* hout = hbuf + (size_t)pw * LBH + (size_t)layer * BH;

      const bf16_t* Bih = Wih_l + (size_t)col0 * RH;  // weight row j = out col
      const bf16_t* Bhh = Whh_l + (size_t)col0 * RH;

      // Unified K in [0, 2H): first H from inp @ W_ih^T, then hprev @ W_hh^T.
      // 32 iterations, each staging a 64x64 A block (double-buffered).
      stage_a(inp, strideA, smem, tid);
      v8f acc = {};
      int buf = 0;
      for (int it = 0; it < 32; ++it) {
        wait_async0();
        __syncthreads();
        if (it + 1 < 32) {
          const int kn = (it + 1) * 64;
          const bf16_t* ga = (kn < RH) ? inp + kn : hprev + (kn - RH);
          const size_t st  = (kn < RH) ? strideA : (size_t)RH;
          stage_a(ga, st, smem + (buf ^ 1) * LDS_BUF, tid);
        }
        const bf16_t* lb = smem + buf * LDS_BUF;
#pragma unroll
        for (int kc = 0; kc < 64; kc += 32) {
          const int kg = it * 64 + kc;
          const bf16_t* Brow = (kg < RH) ? (Bih + kg) : (Bhh + (kg - RH));
          __builtin_prefetch(Brow + 512, 0, 0);       // global_prefetch_b8
          v16bf a = lds_frag(lb, m0, kc, lane);       // ds_load_b128 x2
          v16bf b = glb_frag(Brow, RH, lane);         // global_load_b128 x2
          acc = __builtin_amdgcn_wmma_f32_16x16x32_bf16(
                    false, a, false, b, (short)0, acc, false, false);
        }
        buf ^= 1;
      }

      // Epilogue: D layout: lane<16 -> M=i, lane>=16 -> M=8+i (ISA 7.12.2).
      bf16_t* orow = hout + (size_t)(col0 + ncol);
#pragma unroll
      for (int i = 0; i < 8; ++i) {
        const int m = m0 + i + 8 * hhalf;
        float v = fast_tanh(acc[i] + bcol);
        orow[(size_t)m * RH] = (bf16_t)v;
      }
    }
    grid_sync(&sync[0], &sync[1]);
  }
}

// --------------------------- FC epilogue -----------------------------------
__global__ void rnn_fc(const bf16_t* __restrict__ htop,
                       const float* __restrict__ wfc,
                       const float* __restrict__ bfc,
                       float* __restrict__ out) {
  __shared__ float red[256];
  const int b = blockIdx.x;
  float s = 0.f;
  for (int k = threadIdx.x; k < RH; k += 256)
    s += (float)htop[(size_t)b * RH + k] * wfc[k];
  red[threadIdx.x] = s;
  __syncthreads();
  for (int off = 128; off > 0; off >>= 1) {
    if (threadIdx.x < off) red[threadIdx.x] += red[threadIdx.x + off];
    __syncthreads();
  }
  if (threadIdx.x == 0) out[b] = red[0] + bfc[0];
}

// --------------------------- launcher --------------------------------------
extern "C" void kernel_launch(void* const* d_in, const int* in_sizes, int n_in,
                              void* d_out, int out_size, void* d_ws, size_t ws_size,
                              hipStream_t stream) {
  const float* x    = (const float*)d_in[0];
  const float* W_ih = (const float*)d_in[1];
  const float* W_hh = (const float*)d_in[2];
  const float* b_ih = (const float*)d_in[3];
  const float* b_hh = (const float*)d_in[4];
  const float* W_fc = (const float*)d_in[5];
  const float* b_fc = (const float*)d_in[6];
  float* out = (float*)d_out;

  uint8_t* ws = (uint8_t*)d_ws;
  size_t off = 0;
  auto alloc = [&](size_t bytes) -> void* {
    void* p = ws + off;
    off = (off + bytes + 255) & ~(size_t)255;
    return p;
  };
  bf16_t*   wbih = (bf16_t*)alloc((size_t)RL * RH * RH * sizeof(bf16_t)); // 8 MB
  bf16_t*   wbhh = (bf16_t*)alloc((size_t)RL * RH * RH * sizeof(bf16_t)); // 8 MB
  float*    bias = (float*) alloc((size_t)RL * RH * sizeof(float));       // 16 KB
  bf16_t*   xb   = (bf16_t*)alloc((size_t)RB * RT * RH * sizeof(bf16_t)); // 32 MB
  bf16_t*   hbuf = (bf16_t*)alloc((size_t)2 * LBH * sizeof(bf16_t));      // 1 MB
  unsigned* sync = (unsigned*)alloc(2 * sizeof(unsigned));

  {
    size_t n = (size_t)RL * RH * RH;
    rnn_prep_weights<<<(unsigned)((n + 255) / 256), 256, 0, stream>>>(
        W_ih, W_hh, b_ih, b_hh, wbih, wbhh, bias);
  }
  {
    size_t n = (size_t)RB * RT * RH;
    rnn_prep_x<<<(unsigned)((n + 255) / 256), 256, 0, stream>>>(x, xb);
  }
  {
    size_t n = (size_t)2 * LBH * sizeof(bf16_t) / sizeof(unsigned);
    rnn_zero_ws<<<(unsigned)((n + 255) / 256), 256, 0, stream>>>(
        (unsigned*)hbuf, sync);
  }
  rnn_persistent<<<NWG, 256, 0, stream>>>(xb, wbih, wbhh, bias, hbuf, sync);

  const int pfin = (RT + RL - 2) & 1;  // parity of layer-3 write at t=T-1
  const bf16_t* htop = hbuf + (size_t)pfin * LBH + (size_t)(RL - 1) * BH;
  rnn_fc<<<RB, 256, 0, stream>>>(htop, W_fc, b_fc, out);
}
